// AttentionTorch_62182536511494
// MI455X (gfx1250) — matrix operand order repaired and checked by reading, not compiled, not run on hardware
//
#include <hip/hip_runtime.h>
#include <hip/hip_bf16.h>

// ---------------------------------------------------------------------------
// AlphaFold pair attention, fused for MI455X (gfx1250, wave32, WMMA f16).
//   N_TOKEN=384, C=128, H=4, DH=32
// Pipeline: LN(+pair-bias) -> q/k/v/gate GEMMs (WMMA) -> flash attention
// (WMMA for QK^T and PV) -> gated output GEMM (WMMA, f32 out).
// ---------------------------------------------------------------------------

typedef __attribute__((ext_vector_type(16))) _Float16 v16h;
typedef __attribute__((ext_vector_type(8)))  _Float16 v8h;
typedef __attribute__((ext_vector_type(4)))  _Float16 v4h;
typedef __attribute__((ext_vector_type(8)))  float    v8f;

#define NT   384
#define CC   128
#define NH   4
#define DH   32
#define NROW (NT * NT)          // 147456 rows of [C]
#define QSCALE 0.17677669529663687f  // 1/sqrt(32)

// Load a 16x32 f16 A-fragment (per ISA 7.12.2 16-bit A layout):
//   lanes 0-15 : row r, K = k0+0..7 (v0-3) and k0+16..23 (v4-7)
//   lanes 16-31: row r, K = k0+8..15 and k0+24..31
// caller passes p = rowbase + k0 + half*8 ; we read p[0..7] and p[16..23].
__device__ __forceinline__ v16h load_a16(const _Float16* p) {
    v8h lo = *(const v8h*)p;
    v8h hi = *(const v8h*)(p + 16);
    v16h a;
#pragma unroll
    for (int i = 0; i < 8; ++i) { a[i] = lo[i]; a[i + 8] = hi[i]; }
    return a;
}

__device__ __forceinline__ float wred_sum32(float v) {
#pragma unroll
    for (int d = 16; d >= 1; d >>= 1) v += __shfl_xor(v, d, 32);
    return v;
}
__device__ __forceinline__ float hred_max16(float v) {
#pragma unroll
    for (int d = 8; d >= 1; d >>= 1) v = fmaxf(v, __shfl_xor(v, d, 32));
    return v;
}
__device__ __forceinline__ float hred_sum16(float v) {
#pragma unroll
    for (int d = 8; d >= 1; d >>= 1) v += __shfl_xor(v, d, 32);
    return v;
}

// ---------------------------------------------------------------------------
// Kernel 1: LayerNorm (f32 stats) -> x (f16), plus pair bias [H,N,N] (f32).
// One wave per (i,j) row; 4 channels per lane.
// ---------------------------------------------------------------------------
__global__ void __launch_bounds__(256)
ln_bias_kernel(const float* __restrict__ pair, const float* __restrict__ ln_w,
               const float* __restrict__ ln_b, const float* __restrict__ w_bias,
               _Float16* __restrict__ Xh, float* __restrict__ Bias)
{
    const int lane = threadIdx.x & 31;
    const int wave = threadIdx.x >> 5;
    const int row  = blockIdx.x * 8 + wave;      // 0 .. 147455

    const float4 v = *(const float4*)(pair + (long)row * CC + lane * 4);
    float s  = v.x + v.y + v.z + v.w;
    float ss = v.x * v.x + v.y * v.y + v.z * v.z + v.w * v.w;
    s  = wred_sum32(s);
    ss = wred_sum32(ss);
    const float mu   = s * (1.0f / CC);
    const float var  = ss * (1.0f / CC) - mu * mu;
    const float rstd = rsqrtf(var + 1e-5f);

    const float4 w = *(const float4*)(ln_w + lane * 4);
    const float4 b = *(const float4*)(ln_b + lane * 4);
    float x0 = (v.x - mu) * rstd * w.x + b.x;
    float x1 = (v.y - mu) * rstd * w.y + b.y;
    float x2 = (v.z - mu) * rstd * w.z + b.z;
    float x3 = (v.w - mu) * rstd * w.w + b.w;

    v4h xo; xo[0] = (_Float16)x0; xo[1] = (_Float16)x1;
            xo[2] = (_Float16)x2; xo[3] = (_Float16)x3;
    *(v4h*)(Xh + (long)row * CC + lane * 4) = xo;

    const int i = row / NT, j = row - i * NT;
#pragma unroll
    for (int hh = 0; hh < NH; ++hh) {
        const float4 wb = *(const float4*)(w_bias + hh * CC + lane * 4);
        float bs = x0 * wb.x + x1 * wb.y + x2 * wb.z + x3 * wb.w;
        bs = wred_sum32(bs);
        if (lane == 0) Bias[((hh * NT) + i) * NT + j] = bs;
    }
}

// ---------------------------------------------------------------------------
// Kernel 2/4: Y = A @ W^T, M=147456, N=K=128, WMMA f16 -> f32 acc.
// Block: 128 rows, 8 waves (16 rows each, all 128 cols => 8 accum tiles).
// W (f32, row-major [out][in]) staged in LDS as f16: Wsh[n*128+k] == B^T frag
// source (contiguous per lane).
// MODE: 0 = *QSCALE f16 (q), 1 = plain f16 (k,v), 3 = sigmoid f16 (gate),
//       4 = f32 out.  GATED folds elementwise gate into A (final proj).
// All four A fragments are preloaded (one global clause), and all eight B
// fragments per K-chunk live in registers so ds_loads overlap the WMMAs.
// ---------------------------------------------------------------------------
template <int MODE, bool GATED>
__global__ void __launch_bounds__(256)
gemm128_kernel(const _Float16* __restrict__ A, const float* __restrict__ Wf,
               const _Float16* __restrict__ G, _Float16* __restrict__ Yh,
               float* __restrict__ Yf)
{
    __shared__ _Float16 Wsh[CC * CC];            // 32 KB
    const int tid  = threadIdx.x;
    const int lane = tid & 31;
    const int wave = tid >> 5;
    const int hf   = lane >> 4;                  // half-wave
    const int r    = lane & 15;

    for (int idx = tid; idx < CC * CC; idx += 256)
        Wsh[idx] = (_Float16)Wf[idx];

    const long mbase = (long)blockIdx.x * 128 + wave * 16;
    const _Float16* arow = A + (mbase + r) * CC + hf * 8;

    // Preload all 4 A fragments (whole 128-wide row per lane half).
    v16h af[4];
#pragma unroll
    for (int c = 0; c < 4; ++c) {
        af[c] = load_a16(arow + c * 32);
        if (GATED) {
            const _Float16* grow = G + (mbase + r) * CC + hf * 8 + c * 32;
            v8h glo = *(const v8h*)grow;
            v8h ghi = *(const v8h*)(grow + 16);
#pragma unroll
            for (int i = 0; i < 8; ++i) { af[c][i] *= glo[i]; af[c][i + 8] *= ghi[i]; }
        }
    }
    __syncthreads();

    v8f acc[8] = {};
#pragma unroll
    for (int k0 = 0; k0 < CC; k0 += 32) {
        v16h bf[8];
#pragma unroll
        for (int t = 0; t < 8; ++t)
            bf[t] = load_a16(&Wsh[(t * 16 + r) * CC + k0 + hf * 8]);
#pragma unroll
        for (int t = 0; t < 8; ++t)
            acc[t] = __builtin_amdgcn_wmma_f32_16x16x32_f16(
                false, af[k0 / 32], false, bf[t], (short)0, acc[t], false, false);
    }

#pragma unroll
    for (int t = 0; t < 8; ++t) {
#pragma unroll
        for (int j = 0; j < 8; ++j) {
            const float y = acc[t][j];
            const long orow = mbase + hf * 8 + j;
            const int  ocol = t * 16 + r;
            if (MODE == 4)      Yf[orow * CC + ocol] = y;
            else if (MODE == 0) Yh[orow * CC + ocol] = (_Float16)(y * QSCALE);
            else if (MODE == 3) Yh[orow * CC + ocol] = (_Float16)(1.0f / (1.0f + __expf(-y)));
            else                Yh[orow * CC + ocol] = (_Float16)y;
        }
    }
}

// ---------------------------------------------------------------------------
// Kernel 3: flash attention per (b, h, 128-query slab).
// LDS: K [m][dh] (24KB), V transposed [dh][m] (24KB), per-wave P tile (8KB).
// ---------------------------------------------------------------------------
__global__ void __launch_bounds__(256)
attn_kernel(const _Float16* __restrict__ Q, const _Float16* __restrict__ K,
            const _Float16* __restrict__ V, const float* __restrict__ Bias,
            const unsigned char* __restrict__ Mask, _Float16* __restrict__ WA)
{
    __shared__ _Float16 ksh[NT * DH];        // [m][dh]
    __shared__ _Float16 vsh[DH * NT];        // [dh][m]  (transposed)
    __shared__ _Float16 pbuf[8 * 16 * 32];   // per-wave P tile

    const int b = blockIdx.x, h = blockIdx.y, qb = blockIdx.z;
    const int tid  = threadIdx.x;
    const int lane = tid & 31;
    const int wave = tid >> 5;
    const int hf   = lane >> 4;
    const int r    = lane & 15;

    // stage K and V (per head slice, contiguous 32 f16 per row in source)
    const _Float16* kg = K + ((long)b * NT) * CC + h * DH;
    const _Float16* vg = V + ((long)b * NT) * CC + h * DH;
    for (int idx = tid; idx < NT * 4; idx += 256) {
        const int m = idx >> 2, c = (idx & 3) * 8;
        *(v8h*)(&ksh[m * DH + c]) = *(const v8h*)(kg + (long)m * CC + c);
        v8h vv = *(const v8h*)(vg + (long)m * CC + c);
#pragma unroll
        for (int e = 0; e < 8; ++e) vsh[(c + e) * NT + m] = vv[e];   // transpose
    }
    __syncthreads();

    const int qbase = qb * 128 + wave * 16;
    const v16h qf = load_a16(Q + ((long)b * NT + qbase + r) * CC + h * DH + hf * 8);

    v8f oacc[2] = {};
    float rmax[8], rsum[8];
#pragma unroll
    for (int j = 0; j < 8; ++j) { rmax[j] = -3.0e38f; rsum[j] = 0.0f; }

    const float* biasrow = Bias + ((long)h * NT + qbase) * NT;
    _Float16* pw = pbuf + wave * 512;

    for (int m0 = 0; m0 < NT; m0 += 32) {
        // hoist V fragments: their ds_loads overlap the softmax VALU work
        v16h vf[2];
#pragma unroll
        for (int t = 0; t < 2; ++t)
            vf[t] = load_a16(&vsh[(t * 16 + r) * NT + m0 + hf * 8]);

#pragma unroll
        for (int sub = 0; sub < 2; ++sub) {
            const int m = m0 + sub * 16;
            const v16h kf = load_a16(&ksh[(m + r) * DH + hf * 8]);
            v8f s = __builtin_amdgcn_wmma_f32_16x16x32_f16(
                false, qf, false, kf, (short)0, (v8f){}, false, false);

            const int key = m + r;
            const bool mk = Mask[(long)b * NT + key] != 0;
#pragma unroll
            for (int j = 0; j < 8; ++j) {
                const float sv = s[j] + biasrow[(hf * 8 + j) * NT + key];
                s[j] = mk ? sv : -1e9f;
            }
#pragma unroll
            for (int j = 0; j < 8; ++j) {
                const float tmax = hred_max16(s[j]);          // rowwise (16 lanes)
                const float nm   = fmaxf(rmax[j], tmax);
                const float cf   = __expf(rmax[j] - nm);
                const float pj   = __expf(s[j] - nm);
                rsum[j] = rsum[j] * cf + hred_sum16(pj);
                rmax[j] = nm;
                oacc[0][j] *= cf;
                oacc[1][j] *= cf;
                // scatter P into A-fragment-friendly [row][32] layout
                pw[(hf * 8 + j) * 32 + sub * 16 + r] = (_Float16)pj;
            }
        }
        // P (16x32) @ V (32 x 32dh): per-wave LDS is hardware in-order, no barrier
        const v16h pf = load_a16(pw + r * 32 + hf * 8);
#pragma unroll
        for (int t = 0; t < 2; ++t)
            oacc[t] = __builtin_amdgcn_wmma_f32_16x16x32_f16(
                false, pf, false, vf[t], (short)0, oacc[t], false, false);
    }

    _Float16* wout = WA + ((long)b * NT + qbase) * CC + h * DH;
#pragma unroll
    for (int t = 0; t < 2; ++t) {
#pragma unroll
        for (int j = 0; j < 8; ++j) {
            const float y = oacc[t][j] / rsum[j];
            wout[(hf * 8 + j) * CC + t * 16 + r] = (_Float16)y;
        }
    }
}

// ---------------------------------------------------------------------------
extern "C" void kernel_launch(void* const* d_in, const int* in_sizes, int n_in,
                              void* d_out, int out_size, void* d_ws, size_t ws_size,
                              hipStream_t stream) {
    const float*         pair   = (const float*)d_in[0];
    const unsigned char* mask   = (const unsigned char*)d_in[1];
    const float*         ln_w   = (const float*)d_in[2];
    const float*         ln_b   = (const float*)d_in[3];
    const float*         w_bias = (const float*)d_in[4];
    const float*         wq     = (const float*)d_in[5];
    const float*         wk     = (const float*)d_in[6];
    const float*         wv     = (const float*)d_in[7];
    const float*         wg     = (const float*)d_in[8];
    const float*         wo     = (const float*)d_in[9];
    float*               out    = (float*)d_out;

    const size_t XE = (size_t)NROW * CC;         // 18,874,368 f16 elements
    _Float16* Xh = (_Float16*)d_ws;
    _Float16* Qh = Xh + XE;
    _Float16* Kh = Qh + XE;
    _Float16* Vh = Kh + XE;
    _Float16* Gh = Vh + XE;
    _Float16* Wa = Gh + XE;
    float*  Bias = (float*)(Wa + XE);            // [H, N, N] f32

    ln_bias_kernel<<<NROW / 8, 256, 0, stream>>>(pair, ln_w, ln_b, w_bias, Xh, Bias);

    gemm128_kernel<0, false><<<NROW / 128, 256, 0, stream>>>(Xh, wq, nullptr, Qh, nullptr);
    gemm128_kernel<1, false><<<NROW / 128, 256, 0, stream>>>(Xh, wk, nullptr, Kh, nullptr);
    gemm128_kernel<1, false><<<NROW / 128, 256, 0, stream>>>(Xh, wv, nullptr, Vh, nullptr);
    gemm128_kernel<3, false><<<NROW / 128, 256, 0, stream>>>(Xh, wg, nullptr, Gh, nullptr);

    attn_kernel<<<dim3(NT, NH, 3), 256, 0, stream>>>(Qh, Kh, Vh, Bias, mask, Wa);

    gemm128_kernel<4, true><<<NROW / 128, 256, 0, stream>>>(Wa, wo, Gh, nullptr, out);
}
